// WeightAverage_29368986370494
// MI455X (gfx1250) — compile-verified, matrix operand order checked by reading, not run on hardware
//
#include <hip/hip_runtime.h>
#include <hip/hip_bf16.h>

// ---------------- types ----------------
typedef __attribute__((ext_vector_type(16))) __bf16 v16bf;
typedef __attribute__((ext_vector_type(8)))  float  v8f;

union Frag16 {
    v16bf v;
    uint4 q[2];
    unsigned short u[16];
};

// ---------------- problem constants ----------------
#define BATCH 8
#define CIN   256
#define CO    128
#define HH    96
#define WW    96
#define HWPIX (HH*WW)              // 9216
#define NPIX  (BATCH*HWPIX)        // 73728
#define PTILES (NPIX/16)           // 4608
#define EPSV  1e-8f

// float -> bf16, round-to-nearest-even
__device__ __forceinline__ unsigned short f2bf(float f) {
    unsigned int u = __float_as_uint(f);
    unsigned int r = u + 0x7FFFu + ((u >> 16) & 1u);
    return (unsigned short)(r >> 16);
}

__device__ __forceinline__ float wave_sum32(float v) {
#pragma unroll
    for (int m = 16; m >= 1; m >>= 1) v += __shfl_xor(v, m, 32);
    return v;
}

// =====================================================================
// Kernel 0: pack weights into bf16 WMMA B-fragment layout.
// B-frag (16x16x32 bf16): lane L -> column n = base + (L&15);
//   kbase = kstep*32 + (L>=16 ? 8 : 0);
//   halves j=0..7  -> K = kbase + j
//   halves j=8..15 -> K = kbase + 16 + (j-8)
// packA: [proj(3)][ntile(8)][kstep(8)][lane(32)][16]   (proj GEMMs, K=256)
// packB: [ntile(16)][kstep(4)][lane(32)][16]           (back GEMM,  K=128)
// =====================================================================
__global__ void pack_weights_kernel(const float* __restrict__ wt,
                                    const float* __restrict__ wp,
                                    const float* __restrict__ wg,
                                    const float* __restrict__ wb,
                                    unsigned short* __restrict__ packA,
                                    unsigned short* __restrict__ packB) {
    int idx = blockIdx.x * blockDim.x + threadIdx.x;  // 131072 total
    const int NA = 3 * 8 * 8 * 32 * 16;               // 98304
    if (idx < NA) {
        int j    = idx & 15;
        int lane = (idx >> 4) & 31;
        int ks   = (idx >> 9) & 7;
        int nt   = (idx >> 12) & 7;
        int proj = idx >> 15;
        int n = nt * 16 + (lane & 15);
        int kbase = ks * 32 + ((lane >> 4) << 3);
        int k = (j < 8) ? (kbase + j) : (kbase + 16 + (j - 8));
        const float* src = (proj == 0) ? wt : (proj == 1) ? wp : wg;
        packA[idx] = f2bf(src[n * CIN + k]);
    } else {
        int e = idx - NA;                             // 32768 entries
        if (e >= 16 * 4 * 32 * 16) return;
        int j    = e & 15;
        int lane = (e >> 4) & 31;
        int ks   = (e >> 9) & 3;
        int nt   = e >> 11;
        int c = nt * 16 + (lane & 15);
        int kbase = ks * 32 + ((lane >> 4) << 3);
        int k = (j < 8) ? (kbase + j) : (kbase + 16 + (j - 8));
        packB[e] = f2bf(wb[c * CO + k]);
    }
}

// =====================================================================
// Kernel 1: theta/phi/g projections. One block = one 16-pixel tile.
// 8 waves; wave w owns output n-tile w (16 of 128 channels) for all 3
// projections. x tile staged in LDS as bf16 (padded rows, ds_read_b128
// A-fragments), 8 K-steps of v_wmma_f32_16x16x32_bf16.
// Outputs pixel-major: buf[pix*128 + ch].
// =====================================================================
__global__ void __launch_bounds__(256)
proj_kernel(const float* __restrict__ x,
            const unsigned short* __restrict__ packA,
            const float* __restrict__ bt, const float* __restrict__ bp,
            const float* __restrict__ bg,
            float* __restrict__ thetaB, float* __restrict__ phiB,
            float* __restrict__ gB) {
    __shared__ __align__(16) unsigned short tA[16 * 264];  // 16 pixels x 256ch (+8 pad)

    const int pt   = blockIdx.x;          // pixel tile
    const int tid  = threadIdx.x;
    const int lane = tid & 31;
    const int wid  = tid >> 5;            // wave id = n-tile

    // ---- stage x tile into LDS as bf16 (thread = channel) ----
    {
        const int c  = tid;               // 0..255
        const int b  = (pt * 16) / HWPIX; // HWPIX % 16 == 0
        const int pi = (pt * 16) % HWPIX;
        const float* xp = x + ((size_t)b * CIN + c) * HWPIX + pi;
        float vals[16];
        const float4* xp4 = (const float4*)xp;
#pragma unroll
        for (int q = 0; q < 4; ++q) {
            float4 v = xp4[q];
            vals[q * 4 + 0] = v.x; vals[q * 4 + 1] = v.y;
            vals[q * 4 + 2] = v.z; vals[q * 4 + 3] = v.w;
        }
#pragma unroll
        for (int p = 0; p < 16; ++p) tA[p * 264 + c] = f2bf(vals[p]);
    }
    __syncthreads();

    const int m    = lane & 15;           // A row = pixel within tile
    const int koff = (lane >> 4) << 3;    // +8 K for upper half-lanes

    v8f accT = {}; v8f accP = {}; v8f accG = {};

#pragma unroll
    for (int ks = 0; ks < 8; ++ks) {
        const int kbase = ks * 32 + koff;
        Frag16 fa;
        fa.q[0] = *(const uint4*)&tA[m * 264 + kbase];
        fa.q[1] = *(const uint4*)&tA[m * 264 + kbase + 16];

        Frag16 fbT, fbP, fbG;
        const unsigned short* wbase =
            packA + ((((size_t)0 * 8 + wid) * 8 + ks) * 32 + lane) * 16;
        const size_t projStride = (size_t)8 * 8 * 32 * 16;
        fbT.q[0] = ((const uint4*)wbase)[0];
        fbT.q[1] = ((const uint4*)wbase)[1];
        fbP.q[0] = ((const uint4*)(wbase + projStride))[0];
        fbP.q[1] = ((const uint4*)(wbase + projStride))[1];
        fbG.q[0] = ((const uint4*)(wbase + 2 * projStride))[0];
        fbG.q[1] = ((const uint4*)(wbase + 2 * projStride))[1];

        accT = __builtin_amdgcn_wmma_f32_16x16x32_bf16(false, fa.v, false, fbT.v,
                                                       (short)0, accT, false, false);
        accP = __builtin_amdgcn_wmma_f32_16x16x32_bf16(false, fa.v, false, fbP.v,
                                                       (short)0, accP, false, false);
        accG = __builtin_amdgcn_wmma_f32_16x16x32_bf16(false, fa.v, false, fbG.v,
                                                       (short)0, accG, false, false);
    }

    // ---- store D-fragments; lane -> channel n (fixed), rows -> pixels ----
    const int n   = wid * 16 + (lane & 15);
    const float biasT = bt[n], biasP = bp[n], biasG = bg[n];
    const int prow = pt * 16 + ((lane >> 4) << 3);  // pixel of VGPR row 0
#pragma unroll
    for (int r = 0; r < 8; ++r) {
        const size_t o = (size_t)(prow + r) * CO + n;
        thetaB[o] = accT[r] + biasT;
        phiB[o]   = accP[r] + biasP;
        gB[o]     = accG[r] + biasG;
    }
}

// =====================================================================
// Kernel 2: cosine-similarity attention + weighted average.
// One wave per pixel; 4 channels/lane (128 = 32*4); shfl-xor reductions.
// phi/g buffers (75 MB total) are L2-resident, so the 9x re-reads stay
// on-chip.
// =====================================================================
__global__ void __launch_bounds__(256)
attn_kernel(const float* __restrict__ thetaB, const float* __restrict__ phiB,
            const float* __restrict__ gB, float* __restrict__ waB) {
    const int lane = threadIdx.x & 31;
    const int p    = blockIdx.x * 8 + (threadIdx.x >> 5);
    const int c0   = lane * 4;

    const int b   = p / HWPIX;
    const int rem = p - b * HWPIX;
    const int h   = rem / WW;
    const int w   = rem - h * WW;

    const float4 t = *(const float4*)&thetaB[(size_t)p * CO + c0];
    const float tn = wave_sum32(t.x * t.x + t.y * t.y + t.z * t.z + t.w * t.w);
    const float tnorm = __builtin_sqrtf(tn);

    int np[9];
    float logit[9];
#pragma unroll
    for (int nb = 0; nb < 9; ++nb) {
        int dh = nb / 3 - 1, dw = nb % 3 - 1;
        int hh = min(max(h + dh, 0), HH - 1);
        int ww2 = min(max(w + dw, 0), WW - 1);
        np[nb] = b * HWPIX + hh * WW + ww2;
        const float4 pv = *(const float4*)&phiB[(size_t)np[nb] * CO + c0];
        float d  = pv.x * t.x + pv.y * t.y + pv.z * t.z + pv.w * t.w;
        float pn = pv.x * pv.x + pv.y * pv.y + pv.z * pv.z + pv.w * pv.w;
        d  = wave_sum32(d);
        pn = wave_sum32(pn);
        logit[nb] = d / fmaxf(__builtin_sqrtf(pn) * tnorm, EPSV);
    }

    // softmax over the 9 neighbors (replicated in every lane)
    float mx = logit[0];
#pragma unroll
    for (int nb = 1; nb < 9; ++nb) mx = fmaxf(mx, logit[nb]);
    float s = 0.f;
    float e[9];
#pragma unroll
    for (int nb = 0; nb < 9; ++nb) { e[nb] = __builtin_expf(logit[nb] - mx); s += e[nb]; }
    const float inv = 1.f / s;

    float4 acc = make_float4(0.f, 0.f, 0.f, 0.f);
#pragma unroll
    for (int nb = 0; nb < 9; ++nb) {
        const float a = e[nb] * inv;
        const float4 gv = *(const float4*)&gB[(size_t)np[nb] * CO + c0];
        acc.x += a * gv.x; acc.y += a * gv.y; acc.z += a * gv.z; acc.w += a * gv.w;
    }
    *(float4*)&waB[(size_t)p * CO + c0] = acc;
}

// =====================================================================
// Kernel 3: back-projection GEMM (256 <- 128) + bias + residual add.
// One block = one 16-pixel tile; 8 waves x 2 n-tiles each cover C=256.
// K = 128 -> 4 WMMA steps. Stores fused x + x_res into NCHW output as
// two lane-contiguous float4 per n-tile.
// =====================================================================
__global__ void __launch_bounds__(256)
back_kernel(const float* __restrict__ waB,
            const unsigned short* __restrict__ packB,
            const float* __restrict__ x, const float* __restrict__ bback,
            float* __restrict__ out) {
    const int pt   = blockIdx.x;
    const int lane = threadIdx.x & 31;
    const int wid  = threadIdx.x >> 5;

    const int m    = lane & 15;
    const int koff = (lane >> 4) << 3;
    const float* ap = waB + (size_t)(pt * 16 + m) * CO;

    v8f acc0 = {}; v8f acc1 = {};

#pragma unroll
    for (int ks = 0; ks < 4; ++ks) {
        const int kbase = ks * 32 + koff;
        Frag16 fa;
#pragma unroll
        for (int j = 0; j < 8; ++j) fa.u[j]     = f2bf(ap[kbase + j]);
#pragma unroll
        for (int j = 0; j < 8; ++j) fa.u[8 + j] = f2bf(ap[kbase + 16 + j]);

        const int nt0 = wid, nt1 = wid + 8;
        Frag16 fb0, fb1;
        const unsigned short* w0 = packB + (((size_t)nt0 * 4 + ks) * 32 + lane) * 16;
        const unsigned short* w1 = packB + (((size_t)nt1 * 4 + ks) * 32 + lane) * 16;
        fb0.q[0] = ((const uint4*)w0)[0]; fb0.q[1] = ((const uint4*)w0)[1];
        fb1.q[0] = ((const uint4*)w1)[0]; fb1.q[1] = ((const uint4*)w1)[1];

        acc0 = __builtin_amdgcn_wmma_f32_16x16x32_bf16(false, fa.v, false, fb0.v,
                                                       (short)0, acc0, false, false);
        acc1 = __builtin_amdgcn_wmma_f32_16x16x32_bf16(false, fa.v, false, fb1.v,
                                                       (short)0, acc1, false, false);
    }

    // fused residual + bias store (NCHW)
    const int b      = (pt * 16) / HWPIX;
    const int piximg = (pt * 16) % HWPIX + ((lane >> 4) << 3);
#pragma unroll
    for (int t = 0; t < 2; ++t) {
        const int c = (wid + t * 8) * 16 + (lane & 15);
        const float bias = bback[c];
        const v8f& a = t ? acc1 : acc0;
        const size_t base = ((size_t)b * CIN + c) * HWPIX + piximg;
        float4 xin0 = *(const float4*)&x[base];
        float4 xin1 = *(const float4*)&x[base + 4];
        float4 o0 = make_float4(xin0.x + a[0] + bias, xin0.y + a[1] + bias,
                                xin0.z + a[2] + bias, xin0.w + a[3] + bias);
        float4 o1 = make_float4(xin1.x + a[4] + bias, xin1.y + a[5] + bias,
                                xin1.z + a[6] + bias, xin1.w + a[7] + bias);
        *(float4*)&out[base]     = o0;
        *(float4*)&out[base + 4] = o1;
    }
}

// =====================================================================
// launcher
// =====================================================================
extern "C" void kernel_launch(void* const* d_in, const int* in_sizes, int n_in,
                              void* d_out, int out_size, void* d_ws, size_t ws_size,
                              hipStream_t stream) {
    const float* x       = (const float*)d_in[0];
    const float* w_theta = (const float*)d_in[1];
    const float* b_theta = (const float*)d_in[2];
    const float* w_phi   = (const float*)d_in[3];
    const float* b_phi   = (const float*)d_in[4];
    const float* w_g     = (const float*)d_in[5];
    const float* b_g     = (const float*)d_in[6];
    const float* w_back  = (const float*)d_in[7];
    const float* b_back  = (const float*)d_in[8];
    float* out = (float*)d_out;

    char* ws = (char*)d_ws;
    const size_t szPackA = (size_t)3 * 8 * 8 * 32 * 16 * 2;   // 192 KB
    const size_t szPackB = (size_t)16 * 4 * 32 * 16 * 2;      // 64 KB
    const size_t szBuf   = (size_t)NPIX * CO * sizeof(float); // 37.75 MB each

    unsigned short* packA = (unsigned short*)(ws);
    unsigned short* packB = (unsigned short*)(ws + szPackA);
    float* thetaB = (float*)(ws + szPackA + szPackB);
    float* phiB   = (float*)(ws + szPackA + szPackB + szBuf);
    float* gB     = (float*)(ws + szPackA + szPackB + 2 * szBuf);
    float* waB    = (float*)(ws + szPackA + szPackB + 3 * szBuf);

    // 0) pack weights -> bf16 WMMA fragments (131072 entries)
    pack_weights_kernel<<<131072 / 256, 256, 0, stream>>>(
        w_theta, w_phi, w_g, w_back, packA, packB);

    // 1) theta / phi / g projections (WMMA)
    proj_kernel<<<PTILES, 256, 0, stream>>>(
        x, packA, b_theta, b_phi, b_g, thetaB, phiB, gB);

    // 2) neighbor cosine attention + weighted average
    attn_kernel<<<NPIX / 8, 256, 0, stream>>>(thetaB, phiB, gB, waB);

    // 3) back projection + residual (WMMA)
    back_kernel<<<PTILES, 256, 0, stream>>>(waB, packB, x, b_back, out);
}
